// FusionModule_v6_86337432584662
// MI455X (gfx1250) — matrix operand order repaired and checked by reading, not compiled, not run on hardware
//
#include <hip/hip_runtime.h>
#include <hip/hip_bf16.h>

typedef __attribute__((ext_vector_type(16))) _Float16 v16h;
typedef __attribute__((ext_vector_type(8)))  _Float16 v8h;
typedef __attribute__((ext_vector_type(8)))  float    v8f;

#define NR      8192
#define SDIM    512
#define LRELU   0.01f
#define INV_TAU 2.0f     // 1/0.5
#define KTOP    6

// ---------------------------------------------------------------------------
// CDNA5 async global->LDS copy (16B per lane), tracked by ASYNCcnt.
// VDST = LDS byte address VGPR, VADDR = 64-bit global address VGPR pair.
__device__ inline void async_cp16(uint32_t lds_addr, const void* gp) {
  asm volatile("global_load_async_to_lds_b128 %0, %1, off"
               :: "v"(lds_addr), "v"((unsigned long long)(uintptr_t)gp)
               : "memory");
}
__device__ inline uint32_t lds_lo32(const void* p) {
  return (uint32_t)(uintptr_t)p;   // low 32 bits of generic LDS pointer = LDS offset
}

// ---------------------------------------------------------------------------
// WMMA fragment helpers (contiguous k order: element e holds k = 16*(lane/16) + e in both A and B, wave32)
__device__ inline v16h frag_a(const _Float16* base, int ld, int lane) {
  int r  = lane & 15;
  int kb = (lane >> 4) << 4;          // 0 or 16
  const _Float16* p = base + r * ld + kb;
  v16h f;
#pragma unroll
  for (int e = 0; e < 16; ++e) f[e] = p[e];
  return f;
}
__device__ inline v16h frag_b(const _Float16* base, int ld, int lane) {
  int c  = lane & 15;
  int kb = (lane >> 4) << 4;          // 0 or 16
  const _Float16* p = base + c * ld + kb;
  v16h f;
#pragma unroll
  for (int e = 0; e < 16; ++e) f[e] = p[e];
  return f;
}

// ---------------------------------------------------------------------------
// WMMA GEMM:  C[M x Nn] = A[M x K] * B^T[Nn x K]^T + bias
// 128x64 macro-tile, 8 waves, 4 x (16x16) tiles per wave, async double-buffer.
// mode 0: LeakyReLU, f16 out (Oh).  mode 1: plain, f32 out (Of).
__global__ __launch_bounds__(256) void gemm_wmma(
    const _Float16* __restrict__ A, const _Float16* __restrict__ BT,
    const float* __restrict__ bias,
    _Float16* __restrict__ Oh, float* __restrict__ Of,
    int M, int Nn, int K, int mode)
{
  __shared__ _Float16 As[2][128][48];
  __shared__ _Float16 Bs[2][64][48];
  int tid = threadIdx.x, wid = tid >> 5, lane = tid & 31;
  int gm0 = blockIdx.x * 128, gn0 = blockIdx.y * 64;
  int m0 = (wid & 3) * 2, n0 = (wid >> 2) * 2;      // sub-tile coords (x2 each)
  int arow = tid >> 1, akc = (tid & 1) * 16;        // A: 128 rows x 32 k, 32B/thread
  int brow = tid >> 2, bkc = (tid & 3) * 8;         // B:  64 rows x 32 k, 16B/thread
  v8f acc[2][2] = {};
  int NK = K >> 5;

  auto stage = [&](int ks, int p) {
    int k0 = ks << 5;
    const _Float16* ag = &A[(size_t)(gm0 + arow) * K + k0 + akc];
    async_cp16(lds_lo32(&As[p][arow][akc]),     ag);
    async_cp16(lds_lo32(&As[p][arow][akc + 8]), ag + 8);
    async_cp16(lds_lo32(&Bs[p][brow][bkc]),
               &BT[(size_t)(gn0 + brow) * K + k0 + bkc]);
  };

  stage(0, 0);
  for (int ks = 0; ks < NK; ++ks) {
    int p = ks & 1;
    if (ks + 1 < NK) {
      stage(ks + 1, p ^ 1);
      asm volatile("s_wait_asynccnt 0x3" ::: "memory");  // tile ks resident
    } else {
      asm volatile("s_wait_asynccnt 0x0" ::: "memory");
    }
    __syncthreads();
    v16h a0 = frag_a(&As[p][(m0 + 0) * 16][0], 48, lane);
    v16h a1 = frag_a(&As[p][(m0 + 1) * 16][0], 48, lane);
    v16h b0 = frag_b(&Bs[p][(n0 + 0) * 16][0], 48, lane);
    v16h b1 = frag_b(&Bs[p][(n0 + 1) * 16][0], 48, lane);
    acc[0][0] = __builtin_amdgcn_wmma_f32_16x16x32_f16(false, a0, false, b0, (short)0, acc[0][0], false, false);
    acc[0][1] = __builtin_amdgcn_wmma_f32_16x16x32_f16(false, a0, false, b1, (short)0, acc[0][1], false, false);
    acc[1][0] = __builtin_amdgcn_wmma_f32_16x16x32_f16(false, a1, false, b0, (short)0, acc[1][0], false, false);
    acc[1][1] = __builtin_amdgcn_wmma_f32_16x16x32_f16(false, a1, false, b1, (short)0, acc[1][1], false, false);
    __syncthreads();
  }

  int rn = lane & 15, rmo = (lane >> 4) * 8;
#pragma unroll
  for (int mi = 0; mi < 2; ++mi)
#pragma unroll
    for (int ni = 0; ni < 2; ++ni)
#pragma unroll
      for (int v = 0; v < 8; ++v) {
        int gm = gm0 + (m0 + mi) * 16 + v + rmo;
        int gn = gn0 + (n0 + ni) * 16 + rn;
        float h = acc[mi][ni][v] + bias[gn];
        if (mode == 0) { h = h >= 0.f ? h : LRELU * h; Oh[(size_t)gm * Nn + gn] = (_Float16)h; }
        else           { Of[(size_t)gm * Nn + gn] = h; }
      }
}

// ---------------------------------------------------------------------------
// Fused cossim GEMM + streaming per-row top-6 + optional rowsum of exp(s/tau).
// Q, Mm: row-normalized f16 [8192 x 512]. Block owns 64 rows, streams 8192
// cols; async double-buffered staging pipelined across the whole (n,k) space.
__global__ __launch_bounds__(256) void sim_topk(
    const _Float16* __restrict__ Q, const _Float16* __restrict__ Mm,
    float* __restrict__ rowsum, int* __restrict__ topidx, int doSum)
{
  __shared__ _Float16 As[2][64][48];
  __shared__ _Float16 Bs[2][64][48];
  __shared__ float    St[64][65];
  int tid = threadIdx.x, wid = tid >> 5, lane = tid & 31;
  int rb0 = blockIdx.x * 64;
  int msub = wid & 3, nsubA = (wid >> 2) * 2, nsubB = nsubA + 1;
  int srow = tid >> 2, skc = (tid & 3) * 8;

  float tv[KTOP]; int ti[KTOP];
#pragma unroll
  for (int i = 0; i < KTOP; ++i) { tv[i] = -1e30f; ti[i] = 0; }
  float sum = 0.f;

  const int NSTEP = (NR / 64) * (SDIM / 32);   // 128 n-tiles * 16 k-steps
  auto stage = [&](int s, int p) {
    int n0 = (s >> 4) * 64, k0 = (s & 15) * 32;
    async_cp16(lds_lo32(&As[p][srow][skc]),
               &Q [(size_t)(rb0 + srow) * SDIM + k0 + skc]);
    async_cp16(lds_lo32(&Bs[p][srow][skc]),
               &Mm[(size_t)(n0  + srow) * SDIM + k0 + skc]);
  };

  v8f acc0 = {}, acc1 = {};
  stage(0, 0);
  for (int s = 0; s < NSTEP; ++s) {
    int p = s & 1, k = s & 15;
    if (k == 0) { v8f z = {}; acc0 = z; acc1 = z; }
    if (s + 1 < NSTEP) {
      stage(s + 1, p ^ 1);
      asm volatile("s_wait_asynccnt 0x2" ::: "memory");
    } else {
      asm volatile("s_wait_asynccnt 0x0" ::: "memory");
    }
    __syncthreads();
    v16h a  = frag_a(&As[p][msub * 16][0], 48, lane);
    v16h b0 = frag_b(&Bs[p][nsubA * 16][0], 48, lane);
    v16h b1 = frag_b(&Bs[p][nsubB * 16][0], 48, lane);
    acc0 = __builtin_amdgcn_wmma_f32_16x16x32_f16(false, a, false, b0, (short)0, acc0, false, false);
    acc1 = __builtin_amdgcn_wmma_f32_16x16x32_f16(false, a, false, b1, (short)0, acc1, false, false);

    if (k == 15) {                      // finished one 64x64 similarity tile
      int rn = lane & 15, rmo = (lane >> 4) * 8;
#pragma unroll
      for (int v = 0; v < 8; ++v) {
        St[msub * 16 + v + rmo][nsubA * 16 + rn] = acc0[v];
        St[msub * 16 + v + rmo][nsubB * 16 + rn] = acc1[v];
      }
      __syncthreads();
      if (tid < 64) {
        int n0 = (s >> 4) * 64;
        for (int c = 0; c < 64; ++c) {
          float sv = St[tid][c];
          if (doSum) sum += __expf(sv * INV_TAU);
          if (sv > tv[KTOP - 1]) {      // exp monotonic: topk(sm)==topk(cossim)
            float cv = sv; int ci = n0 + c;
#pragma unroll
            for (int i = 0; i < KTOP; ++i) {
              if (cv > tv[i]) {
                float tmv = tv[i]; int tmi = ti[i];
                tv[i] = cv; ti[i] = ci; cv = tmv; ci = tmi;
              }
            }
          }
        }
      }
      __syncthreads();
    } else {
      __syncthreads();
    }
  }
  if (tid < 64) {
    int r = rb0 + tid;
    if (doSum) rowsum[r] = sum;
#pragma unroll
    for (int i = 0; i < KTOP; ++i) topidx[r * KTOP + i] = ti[i];
  }
}

// ---------------------------------------------------------------------------
// Per-column mean / rsqrt(var+eps) over 8192 rows (population variance).
__global__ void colstats(const float* __restrict__ H, float* __restrict__ mu,
                         float* __restrict__ rvar)
{
  int c = blockIdx.x * blockDim.x + threadIdx.x;
  if (c >= SDIM) return;
  float s = 0.f, s2 = 0.f;
  for (int r = 0; r < NR; ++r) { float v = H[(size_t)r * SDIM + c]; s += v; s2 += v * v; }
  float m = s / (float)NR;
  mu[c] = m;
  rvar[c] = rsqrtf(fmaxf(s2 / (float)NR - m * m, 0.f) + 1e-5f);
}

// BN apply: write f32 output row, plus row-L2-normalized f16 copy for cossim.
__global__ __launch_bounds__(256) void bn_apply(
    const float* __restrict__ H, const float* __restrict__ mu,
    const float* __restrict__ rvar, const float* __restrict__ g,
    const float* __restrict__ b, float* __restrict__ Y, _Float16* __restrict__ Yn)
{
  __shared__ float red[256];
  int r = blockIdx.x, t = threadIdx.x, t2 = t + 256;
  size_t base = (size_t)r * SDIM;
  float y0 = (H[base + t ] - mu[t ]) * rvar[t ] * g[t ] + b[t ];
  float y1 = (H[base + t2] - mu[t2]) * rvar[t2] * g[t2] + b[t2];
  Y[base + t] = y0; Y[base + t2] = y1;
  red[t] = y0 * y0 + y1 * y1;
  __syncthreads();
  for (int o = 128; o > 0; o >>= 1) { if (t < o) red[t] += red[t + o]; __syncthreads(); }
  float inv = 1.f / (sqrtf(red[0]) + 1e-12f);
  Yn[base + t]  = (_Float16)(y0 * inv);
  Yn[base + t2] = (_Float16)(y1 * inv);
}

// Row-normalize an f32 [8192 x 512] matrix into f16.
__global__ __launch_bounds__(256) void rownorm(
    const float* __restrict__ X, _Float16* __restrict__ Xn)
{
  __shared__ float red[256];
  int r = blockIdx.x, t = threadIdx.x, t2 = t + 256;
  size_t base = (size_t)r * SDIM;
  float y0 = X[base + t], y1 = X[base + t2];
  red[t] = y0 * y0 + y1 * y1;
  __syncthreads();
  for (int o = 128; o > 0; o >>= 1) { if (t < o) red[t] += red[t + o]; __syncthreads(); }
  float inv = 1.f / (sqrtf(red[0]) + 1e-12f);
  Xn[base + t]  = (_Float16)(y0 * inv);
  Xn[base + t2] = (_Float16)(y1 * inv);
}

__global__ void cvt_f32_f16(const float* __restrict__ X, _Float16* __restrict__ Y, size_t n)
{
  size_t i = (size_t)blockIdx.x * blockDim.x + threadIdx.x;
  size_t stride = (size_t)gridDim.x * blockDim.x;
  for (; i < n; i += stride) Y[i] = (_Float16)X[i];
}

// Transpose-convert: X[K x Nc] f32  ->  Y[Nc x K] f16
__global__ void cvtT_f32_f16(const float* __restrict__ X, _Float16* __restrict__ Y,
                             int K, int Nc)
{
  size_t n = (size_t)K * Nc;
  size_t i = (size_t)blockIdx.x * blockDim.x + threadIdx.x;
  size_t stride = (size_t)gridDim.x * blockDim.x;
  for (; i < n; i += stride) {
    int k = (int)(i / Nc), c = (int)(i % Nc);
    Y[(size_t)c * K + k] = (_Float16)X[i];
  }
}

__global__ void zero1(float* p) { if (threadIdx.x == 0 && blockIdx.x == 0) p[0] = 0.f; }

// ---------------------------------------------------------------------------
// Loss: one wave per row; 36 gathered sm entries recomputed as wave dots.
__global__ __launch_bounds__(256) void loss_kernel(
    const _Float16* __restrict__ Q, const _Float16* __restrict__ Mm,
    const float* __restrict__ rowsum, const int* __restrict__ topS,
    const int* __restrict__ topM, float* __restrict__ out, float invN)
{
  int wid = threadIdx.x >> 5, lane = threadIdx.x & 31;
  int r = blockIdx.x * 8 + wid;
  const _Float16* qp = Q + (size_t)r * SDIM + lane * 16;
  v8h q0 = *(const v8h*)qp, q1 = *(const v8h*)(qp + 8);
  float q[16];
#pragma unroll
  for (int e = 0; e < 8; ++e) { q[e] = (float)q0[e]; q[8 + e] = (float)q1[e]; }
  float pos = 0.f;
  for (int i = 0; i < KTOP; ++i) {
    int a = topS[r * KTOP + i];
#pragma unroll
    for (int j = 0; j < KTOP; ++j) {
      int c = topM[a * KTOP + j];
      const _Float16* mp = Mm + (size_t)c * SDIM + lane * 16;
      v8h m0 = *(const v8h*)mp, m1 = *(const v8h*)(mp + 8);
      float acc = 0.f;
#pragma unroll
      for (int e = 0; e < 8; ++e) acc += q[e] * (float)m0[e] + q[8 + e] * (float)m1[e];
#pragma unroll
      for (int off = 16; off > 0; off >>= 1) acc += __shfl_xor(acc, off, 32);
      pos += __expf(acc * INV_TAU);
    }
  }
  if (lane == 0) atomicAdd(out, (logf(rowsum[r]) - logf(pos)) * invN);
}

// ---------------------------------------------------------------------------
extern "C" void kernel_launch(void* const* d_in, const int* in_sizes, int n_in,
                              void* d_out, int out_size, void* d_ws, size_t ws_size,
                              hipStream_t stream)
{
  (void)in_sizes; (void)n_in; (void)out_size; (void)ws_size;
  const float* stru = (const float*)d_in[0];
  const float* visu = (const float*)d_in[1];
  const float* ling = (const float*)d_in[2];
  const float* Wl1 = (const float*)d_in[3];  const float* bl1 = (const float*)d_in[4];
  const float* Wl2 = (const float*)d_in[5];  const float* bl2 = (const float*)d_in[6];
  const float* gl  = (const float*)d_in[7];  const float* bel = (const float*)d_in[8];
  const float* Wv1 = (const float*)d_in[9];  const float* bv1 = (const float*)d_in[10];
  const float* Wv2 = (const float*)d_in[11]; const float* bv2 = (const float*)d_in[12];
  const float* gv  = (const float*)d_in[13]; const float* bev = (const float*)d_in[14];
  float* out = (float*)d_out;

  const int Nr = NR, Sd = SDIM, Vd = 4096, Ld = 768;
  char* ws = (char*)d_ws; size_t off = 0;
  auto alloc = [&](size_t bytes) -> char* {
    char* p = ws + off; off = (off + bytes + 255) & ~(size_t)255; return p;
  };
  _Float16* Xl  = (_Float16*)alloc((size_t)Nr * Ld * 2);
  _Float16* Xv  = (_Float16*)alloc((size_t)Nr * Vd * 2);
  _Float16* W1T = (_Float16*)alloc((size_t)Sd * Vd * 2);   // reused, max K
  _Float16* W2T = (_Float16*)alloc((size_t)Sd * Sd * 2);
  _Float16* h1  = (_Float16*)alloc((size_t)Nr * Sd * 2);
  float*    h2  = (float*)   alloc((size_t)Nr * Sd * 4);
  float*    mu  = (float*)   alloc(Sd * 4);
  float*    rv  = (float*)   alloc(Sd * 4);
  _Float16* sn  = (_Float16*)alloc((size_t)Nr * Sd * 2);
  _Float16* yn  = (_Float16*)alloc((size_t)Nr * Sd * 2);
  float*    rs  = (float*)   alloc((size_t)Nr * 4);
  int*      tS  = (int*)     alloc((size_t)Nr * KTOP * 4);
  int*      tM  = (int*)     alloc((size_t)Nr * KTOP * 4);

  zero1<<<1, 1, 0, stream>>>(out);
  rownorm<<<Nr, 256, 0, stream>>>(stru, sn);

  auto branch = [&](const float* X, int K1, const float* W1, const float* b1,
                    const float* W2, const float* b2, const float* g,
                    const float* be, _Float16* X16, float* yout) {
    cvt_f32_f16 <<<1024, 256, 0, stream>>>(X, X16, (size_t)Nr * K1);
    cvtT_f32_f16<<<512, 256, 0, stream>>>(W1, W1T, K1, Sd);
    cvtT_f32_f16<<<512, 256, 0, stream>>>(W2, W2T, Sd, Sd);
    dim3 gg(Nr / 128, Sd / 64);
    gemm_wmma<<<gg, 256, 0, stream>>>(X16, W1T, b1, h1, nullptr, Nr, Sd, K1, 0);
    gemm_wmma<<<gg, 256, 0, stream>>>(h1, W2T, b2, nullptr, h2, Nr, Sd, Sd, 1);
    colstats<<<2, 256, 0, stream>>>(h2, mu, rv);
    bn_apply<<<Nr, 256, 0, stream>>>(h2, mu, rv, g, be, yout, yn);
    sim_topk<<<Nr / 64, 256, 0, stream>>>(sn, yn, rs, tS, 1);   // sm:  top6 + rowsum
    sim_topk<<<Nr / 64, 256, 0, stream>>>(yn, yn, rs, tM, 0);   // mmd: top6 only
    loss_kernel<<<Nr / 8, 256, 0, stream>>>(sn, yn, rs, tS, tM, out, 1.0f / Nr);
  };

  branch(ling, Ld, Wl1, bl1, Wl2, bl2, gl, bel, Xl, out + 1);
  branch(visu, Vd, Wv1, bv1, Wv2, bv2, gv, bev, Xv, out + 1 + (size_t)Nr * Sd);
}
